// MyGCNNet_7808250544371
// MI455X (gfx1250) — matrix operand order repaired and checked by reading, not compiled, not run on hardware
//
#include <hip/hip_runtime.h>
#include <hip/hip_bf16.h>
#include <math.h>

// ---------------------------------------------------------------------------
// Types for CDNA5 WMMA
// ---------------------------------------------------------------------------
typedef __bf16 bf16;
typedef __attribute__((ext_vector_type(16))) __bf16 v16bf;
typedef __attribute__((ext_vector_type(8)))  float  v8f;

#define HID 512
#define INDIM 256

// Native f32 -> bf16 conversion (lowers to v_cvt_pk_bf16_f32 on gfx1250).
__device__ __forceinline__ uint4 pack8(const float* f) {
    union { bf16 h[8]; uint4 q; } u;
#pragma unroll
    for (int i = 0; i < 8; ++i) u.h[i] = (bf16)f[i];
    return u.q;
}
struct Q2 { uint4 a, b; };   // 32B == one v16bf fragment

// ---------------------------------------------------------------------------
// WMMA GEMM:  C[M,N] = concat(A0[:, :K0], A1[:, :Ktot-K0]) @ W[Ktot,N] + bias
//
// LDS tiles are stored in *fragment order* so each wave lane fetches its whole
// 16-element bf16 fragment as two ds_load_b128:
//   A (16x32): lane = (m&15) + 16*((k>>3)&1), elem i = (k&7) + 8*((k>>4)&1)
//   B (32x16): lane = (n&15) + 16*((k>>3)&1), elem i = (k&7) + 8*(k>>4)
// Double-buffered: global loads for tile k+1 issue before the WMMAs of tile k.
// All 5 fragments (A + 4xB) are fetched into distinct registers before the
// WMMA chain so DS waits stagger instead of fully draining per WMMA.
// BM=128, BN=64, BK=32; 256 threads = 8 waves; wave w owns m-tile w (16 rows)
// across all 64 cols (4 f32 accumulators).
// ---------------------------------------------------------------------------
#define BM 128
#define BN 64
#define BK 32
#define AQ 512   // uint4 quads per A buffer: 8 tiles * 32 lanes * 2
#define BQ 256   // uint4 quads per B buffer: 4 tiles * 32 lanes * 2

__global__ __launch_bounds__(256) void gemm_bf16_wmma(
    const float* __restrict__ A0, const float* __restrict__ A1,
    const float* __restrict__ W,  const float* __restrict__ bias,
    float* __restrict__ C, int M, int K0, int Ktot, int N)
{
    __shared__ uint4 sAq[2 * AQ];   // 16 KB
    __shared__ uint4 sBq[2 * BQ];   //  8 KB

    const int tid  = threadIdx.x;
    const int wave = tid >> 5;
    const int lane = tid & 31;
    const int lm   = lane & 15;
    const int lh   = lane >> 4;
    const int rowBlk = blockIdx.y * BM;
    const int colBlk = blockIdx.x * BN;

    // A-loader coords: row ar (0..127), 16-col half ac (0/1)
    const int ar = tid >> 1, ac = tid & 1;
    const int atile = ar >> 4, amlane = ar & 15;
    // B-loader coords: column bn (0..63), k-chunk bk0 in {0,8,16,24}
    const int bn = tid & 63, bk0 = (tid >> 6) * 8;
    const int btile = bn >> 4;
    const int blane = (bn & 15) + 16 * ((bk0 >> 3) & 1);
    const int bh    = bk0 >> 4;

    v8f acc[4];
#pragma unroll
    for (int t = 0; t < 4; ++t)
#pragma unroll
        for (int i = 0; i < 8; ++i) acc[t][i] = 0.0f;

    float aReg[16];
    float bReg[8];

    auto loadG = [&](int kt) {
        // A: 16 contiguous fp32 (64B) per thread
        const int kbase = kt + ac * 16;                 // K0 % 16 == 0 -> no crossing
        const float* srcp   = (kbase < K0) ? A0 : A1;
        const int    kcol   = (kbase < K0) ? kbase : (kbase - K0);
        const int    stride = (kbase < K0) ? K0 : (Ktot - K0);
        const int    grow   = rowBlk + ar;
        if (grow < M) {
            const float* p = srcp + (size_t)grow * stride + kcol;
#pragma unroll
            for (int i = 0; i < 16; ++i) aReg[i] = p[i];
        } else {
#pragma unroll
            for (int i = 0; i < 16; ++i) aReg[i] = 0.0f;
        }
        // B: 8-deep column slice; consecutive lanes take consecutive n -> coalesced
        const float* q = W + (size_t)(kt + bk0) * N + colBlk + bn;
#pragma unroll
        for (int j = 0; j < 8; ++j) bReg[j] = q[(size_t)j * N];
    };
    auto storeL = [&](int buf) {
        // A chunk0 (aReg[0..7])  -> lane amlane,    elems [ac*8, ac*8+8)
        // A chunk1 (aReg[8..15]) -> lane amlane+16, elems [ac*8, ac*8+8)
        sAq[buf * AQ + (atile * 32 + amlane)      * 2 + ac] = pack8(&aReg[0]);
        sAq[buf * AQ + (atile * 32 + amlane + 16) * 2 + ac] = pack8(&aReg[8]);
        sBq[buf * BQ + (btile * 32 + blane)       * 2 + bh] = pack8(bReg);
    };

    loadG(0);
    storeL(0);
    __syncthreads();

    const int nkt = Ktot / BK;
    for (int it = 0; it < nkt; ++it) {
        const int buf = it & 1;
        if (it + 1 < nkt) loadG((it + 1) * BK);                 // overlap with WMMAs
        if (it + 2 < nkt)                                        // GL2 prefetch 2 ahead
            __builtin_prefetch(W + (size_t)((it + 2) * BK + bk0) * N + colBlk + bn, 0, 1);

        // ---- fetch all fragments first (distinct regs -> staggered DS waits)
        Q2 qa;
        qa.a = sAq[buf * AQ + (wave * 32 + lane) * 2 + 0];
        qa.b = sAq[buf * AQ + (wave * 32 + lane) * 2 + 1];
        Q2 qb[4];
#pragma unroll
        for (int t = 0; t < 4; ++t) {
            qb[t].a = sBq[buf * BQ + (t * 32 + lane) * 2 + 0];
            qb[t].b = sBq[buf * BQ + (t * 32 + lane) * 2 + 1];
        }
        const v16bf afrag = __builtin_bit_cast(v16bf, qa);
#pragma unroll
        for (int t = 0; t < 4; ++t) {
            const v16bf bfrag = __builtin_bit_cast(v16bf, qb[t]);
            acc[t] = __builtin_amdgcn_wmma_f32_16x16x32_bf16(
                false, afrag, false, bfrag, (short)0, acc[t], false, false);
        }

        if (it + 1 < nkt) storeL(buf ^ 1);
        __syncthreads();
    }

    // ---- epilogue: bias + store (C/D layout: lane -> N=lm, VGPR v -> M=v+lh*8)
#pragma unroll
    for (int t = 0; t < 4; ++t) {
        const int n  = colBlk + t * 16 + lm;
        const float bv = bias ? bias[n] : 0.0f;
#pragma unroll
        for (int v = 0; v < 8; ++v) {
            const int m = rowBlk + wave * 16 + v + lh * 8;
            if (m < M) C[(size_t)m * N + n] = acc[t][v] + bv;
        }
    }
}

// ---------------------------------------------------------------------------
// Elementwise / graph kernels
// ---------------------------------------------------------------------------
__global__ void zero_kernel(float* __restrict__ p, size_t n) {
    size_t i = (size_t)blockIdx.x * blockDim.x + threadIdx.x;
    if (i < n) p[i] = 0.0f;
}

__global__ void deg_kernel(const int* __restrict__ dst, float* __restrict__ deg, int E) {
    int e = blockIdx.x * blockDim.x + threadIdx.x;
    if (e < E) atomicAdd(&deg[dst[e]], 1.0f);
}

// 4 edges per block, 64 lanes per edge covering 512 channels
__global__ __launch_bounds__(256) void scatter_kernel(
    const float* __restrict__ h, const int* __restrict__ src,
    const int* __restrict__ dst, float* __restrict__ aggr, int E)
{
    const int e = blockIdx.x * 4 + (threadIdx.x >> 6);
    if (e >= E) return;
    const int s = src[e], d = dst[e];
    const float* __restrict__ hp = h + (size_t)s * HID;
    float* __restrict__ ap = aggr + (size_t)d * HID;
    for (int c = (threadIdx.x & 63); c < HID; c += 64)
        atomicAdd(&ap[c], hp[c]);
}

__global__ void divdeg_kernel(float* __restrict__ aggr, const float* __restrict__ deg, size_t total) {
    size_t i = (size_t)blockIdx.x * blockDim.x + threadIdx.x;
    if (i < total) {
        float d = deg[i >> 9];            // HID == 512
        aggr[i] /= fmaxf(d, 1.0f);
    }
}

// wave-per-row L2 normalize (F.normalize, clamp at 1e-12)
__global__ __launch_bounds__(256) void rownorm_kernel(float* __restrict__ h, int M) {
    const int row  = blockIdx.x * 8 + (threadIdx.x >> 5);
    const int lane = threadIdx.x & 31;
    if (row >= M) return;
    float* p = h + (size_t)row * HID;
    float ss = 0.0f;
    for (int c = lane; c < HID; c += 32) { float v = p[c]; ss += v * v; }
#pragma unroll
    for (int off = 16; off > 0; off >>= 1) ss += __shfl_down(ss, off, 32);
    ss = __shfl(ss, 0, 32);
    const float inv = 1.0f / fmaxf(sqrtf(ss), 1e-12f);
    for (int c = lane; c < HID; c += 32) p[c] *= inv;
}

// per-column sum / sumsq (biased-var BN, training-mode stats)
__global__ __launch_bounds__(256) void colstats_kernel(
    const float* __restrict__ h, float* __restrict__ sums,
    float* __restrict__ sumsq, int M)
{
    const int c = blockIdx.x * blockDim.x + threadIdx.x;   // 0..511
    float s = 0.0f, ss = 0.0f;
    for (int r = blockIdx.y; r < M; r += gridDim.y) {
        float v = h[(size_t)r * HID + c];
        s += v; ss += v * v;
    }
    atomicAdd(&sums[c], s);
    atomicAdd(&sumsq[c], ss);
}

__global__ void bnrelu_kernel(float* __restrict__ h, const float* __restrict__ sums,
                              const float* __restrict__ sumsq,
                              const float* __restrict__ g, const float* __restrict__ be,
                              int M)
{
    size_t i = (size_t)blockIdx.x * blockDim.x + threadIdx.x;
    if (i >= (size_t)M * HID) return;
    const int c = (int)(i & (HID - 1));
    const float invM = 1.0f / (float)M;
    const float mean = sums[c] * invM;
    const float var  = sumsq[c] * invM - mean * mean;
    const float v = (h[i] - mean) * rsqrtf(var + 1e-5f) * g[c] + be[c];
    h[i] = fmaxf(v, 0.0f);
}

__global__ void addres_kernel(float* __restrict__ h, const float* __restrict__ u, size_t n) {
    size_t i = (size_t)blockIdx.x * blockDim.x + threadIdx.x;
    if (i < n) h[i] += u[i];
}

// attention + scaled segment-sum pooling; wave per row
__global__ __launch_bounds__(256) void attpool_kernel(
    const float* __restrict__ h, const float* __restrict__ Watt,
    const int* __restrict__ batch, float* __restrict__ pool,
    float* __restrict__ cnt, int M)
{
    const int row  = blockIdx.x * 8 + (threadIdx.x >> 5);
    const int lane = threadIdx.x & 31;
    if (row >= M) return;
    const float* p = h + (size_t)row * HID;
    float dot = 0.0f;
    for (int c = lane; c < HID; c += 32) dot += p[c] * Watt[c];
#pragma unroll
    for (int off = 16; off > 0; off >>= 1) dot += __shfl_down(dot, off, 32);
    dot = __shfl(dot, 0, 32);
    const float att = 1.0f / (1.0f + expf(-dot));
    const float sc  = (att + 1.0f) * 0.5f;       // (att*h + h)/2
    const int b = batch[row];
    float* bp = pool + (size_t)b * HID;
    for (int c = lane; c < HID; c += 32) atomicAdd(&bp[c], p[c] * sc);
    if (lane == 0) atomicAdd(&cnt[b], 1.0f);
}

__global__ void pooldiv_kernel(float* __restrict__ pool, const float* __restrict__ cnt, size_t total) {
    size_t i = (size_t)blockIdx.x * blockDim.x + threadIdx.x;
    if (i < total) pool[i] /= fmaxf(cnt[i >> 9], 1.0f);
}

// readout: out[b, j] = sum_c hg[b, c] * Wr[c, j]; hg row cached in LDS
__global__ __launch_bounds__(256) void readout_kernel(
    const float* __restrict__ hg, const float* __restrict__ Wr,
    float* __restrict__ out, int NC)
{
    __shared__ float sh[HID];
    const int b = blockIdx.y;
    for (int c = threadIdx.x; c < HID; c += 256) sh[c] = hg[(size_t)b * HID + c];
    __syncthreads();
    const int j = blockIdx.x * 256 + threadIdx.x;
    if (j >= NC) return;
    float acc = 0.0f;
    for (int c = 0; c < HID; ++c) acc = fmaf(sh[c], Wr[(size_t)c * NC + j], acc);
    out[(size_t)b * NC + j] = acc;
}

// ---------------------------------------------------------------------------
// Orchestration
// ---------------------------------------------------------------------------
extern "C" void kernel_launch(void* const* d_in, const int* in_sizes, int n_in,
                              void* d_out, int out_size, void* d_ws, size_t ws_size,
                              hipStream_t stream)
{
    (void)n_in; (void)ws_size;
    const float* x       = (const float*)d_in[0];
    const int*   eidx    = (const int*)d_in[1];
    const int*   batch   = (const int*)d_in[2];
    const float* W_embed = (const float*)d_in[3];
    const float* W1      = (const float*)d_in[4];
    const float* b1      = (const float*)d_in[5];
    const float* g1      = (const float*)d_in[6];
    const float* be1     = (const float*)d_in[7];
    const float* W2      = (const float*)d_in[8];
    const float* b2      = (const float*)d_in[9];
    const float* g2      = (const float*)d_in[10];
    const float* be2     = (const float*)d_in[11];
    const float* W_att   = (const float*)d_in[12];
    const float* W_read  = (const float*)d_in[13];
    float* out = (float*)d_out;

    const int Nn = in_sizes[0] / INDIM;   // 60000
    const int E  = in_sizes[1] / 2;       // 360000
    const int NC = 1000;
    const int NB = out_size / NC;         // 64
    const int* src = eidx;
    const int* dst = eidx + E;

    // workspace carve-out
    char* ws = (char*)d_ws;
    size_t off = 0;
    auto take = [&](size_t bytes) -> float* {
        off = (off + 255) & ~(size_t)255;
        float* p = (float*)(ws + off);
        off += bytes;
        return p;
    };
    const size_t feat_b = (size_t)Nn * HID * sizeof(float);
    float* h     = take(feat_b);
    float* t     = take(feat_b);
    float* u     = take(feat_b);
    float* aggr  = take(feat_b);
    float* deg   = take((size_t)Nn * sizeof(float));
    float* sums  = take(HID * sizeof(float));
    float* sumsq = take(HID * sizeof(float));
    float* pool  = take((size_t)NB * HID * sizeof(float));
    float* cnt   = take((size_t)NB * sizeof(float));

    const size_t feat_n = (size_t)Nn * HID;
    auto zero = [&](float* p, size_t n) {
        zero_kernel<<<(unsigned)((n + 255) / 256), 256, 0, stream>>>(p, n);
    };
    const dim3 ggrid(HID / BN, (unsigned)((Nn + BM - 1) / BM));   // (8, 469)
    const unsigned rowgrid = (unsigned)((Nn + 7) / 8);
    const unsigned elgrid  = (unsigned)((feat_n + 255) / 256);
    const unsigned edgegrid = (unsigned)((E + 3) / 4);

    // 1) embed: h = x @ W_embed
    gemm_bf16_wmma<<<ggrid, 256, 0, stream>>>(x, nullptr, W_embed, nullptr,
                                              h, Nn, INDIM, INDIM, HID);
    // 2) degree counts (same for every conv)
    zero(deg, (size_t)Nn);
    deg_kernel<<<(E + 255) / 256, 256, 0, stream>>>(dst, deg, E);

    // 3) three residual SAGE blocks
    for (int l = 0; l < 3; ++l) {
        // ---- conv 1: t = relu(bn(normalize([h | mean_agg(h)] @ W1 + b1)))
        zero(aggr, feat_n);
        scatter_kernel<<<edgegrid, 256, 0, stream>>>(h, src, dst, aggr, E);
        divdeg_kernel<<<elgrid, 256, 0, stream>>>(aggr, deg, feat_n);
        gemm_bf16_wmma<<<ggrid, 256, 0, stream>>>(h, aggr,
            W1 + (size_t)l * 2 * HID * HID, b1 + (size_t)l * HID,
            t, Nn, HID, 2 * HID, HID);
        rownorm_kernel<<<rowgrid, 256, 0, stream>>>(t, Nn);
        zero(sums, HID); zero(sumsq, HID);
        colstats_kernel<<<dim3(2, 128), 256, 0, stream>>>(t, sums, sumsq, Nn);
        bnrelu_kernel<<<elgrid, 256, 0, stream>>>(t, sums, sumsq,
            g1 + (size_t)l * HID, be1 + (size_t)l * HID, Nn);

        // ---- conv 2: u = relu(bn(normalize([t | mean_agg(t)] @ W2 + b2)))
        zero(aggr, feat_n);
        scatter_kernel<<<edgegrid, 256, 0, stream>>>(t, src, dst, aggr, E);
        divdeg_kernel<<<elgrid, 256, 0, stream>>>(aggr, deg, feat_n);
        gemm_bf16_wmma<<<ggrid, 256, 0, stream>>>(t, aggr,
            W2 + (size_t)l * 2 * HID * HID, b2 + (size_t)l * HID,
            u, Nn, HID, 2 * HID, HID);
        rownorm_kernel<<<rowgrid, 256, 0, stream>>>(u, Nn);
        zero(sums, HID); zero(sumsq, HID);
        colstats_kernel<<<dim3(2, 128), 256, 0, stream>>>(u, sums, sumsq, Nn);
        bnrelu_kernel<<<elgrid, 256, 0, stream>>>(u, sums, sumsq,
            g2 + (size_t)l * HID, be2 + (size_t)l * HID, Nn);

        // ---- residual
        addres_kernel<<<elgrid, 256, 0, stream>>>(h, u, feat_n);
    }

    // 4) attention + mean pooling + readout
    zero(pool, (size_t)NB * HID);
    zero(cnt, (size_t)NB);
    attpool_kernel<<<rowgrid, 256, 0, stream>>>(h, W_att, batch, pool, cnt, Nn);
    pooldiv_kernel<<<((size_t)NB * HID + 255) / 256, 256, 0, stream>>>(pool, cnt, (size_t)NB * HID);
    readout_kernel<<<dim3((NC + 255) / 256, NB), 256, 0, stream>>>(pool, W_read, out, NC);
}
